// GNNPlusImproved_57260503990724
// MI455X (gfx1250) — compile-verified, matrix operand order checked by reading, not run on hardware
//
#include <hip/hip_runtime.h>

typedef __attribute__((ext_vector_type(16))) _Float16 v16h;
typedef __attribute__((ext_vector_type(8)))  float    v8f;

#define N_NODES  50000
#define N_EDGES  800000
#define N_GRAPHS 512
#define NODE_IN  128
#define EDGE_IN  64
#define HIDDEN   128
#define SHARED   64

#if __has_builtin(__builtin_amdgcn_global_load_async_to_lds_b128) && \
    __has_builtin(__builtin_amdgcn_s_wait_asynccnt)
#define HAS_ASYNC_LDS 1
#else
#define HAS_ASYNC_LDS 0
#endif

// K index covered by half t of a lane (hi = lane>>4) for 16-bit A/B fragments
// (ISA 7.12.2: lanes 0-15 hold K=0..7 then 16..23; lanes 16-31 hold K=8..15 then 24..31)
__device__ __forceinline__ int kmap(int t, int hi) {
  return t + 8 * hi + ((t >= 8) ? 8 : 0);
}

__device__ __forceinline__ v8f wmma_f16(v16h a, v16h b, v8f c) {
  return __builtin_amdgcn_wmma_f32_16x16x32_f16(false, a, false, b, (short)0, c,
                                                false, false);
}

// ---------------------------------------------------------------------------
// Pack a row-major f32 weight [K x N] into WMMA-B fragment order (f16):
//   P[((nt*(K/32)+kk)*32 + lane)*16 + t] = W[(kk*32 + kmap(t,lane>>4))*N + nt*16 + (lane&15)]
// so a B fragment is one contiguous 32-byte v16h load per lane.
// ---------------------------------------------------------------------------
__global__ void pack_w_kernel(const float* __restrict__ W, _Float16* __restrict__ P,
                              int K, int N) {
  int i = blockIdx.x * blockDim.x + threadIdx.x;
  if (i >= K * N) return;
  int t    = i & 15;
  int lane = (i >> 4) & 31;
  int rest = i >> 9;
  int kk   = rest % (K / 32);
  int nt   = rest / (K / 32);
  int hi = lane >> 4, n = lane & 15;
  int k = kk * 32 + kmap(t, hi);
  P[i] = (_Float16)W[k * N + nt * 16 + n];
}

__global__ void copy4_kernel(const float4* __restrict__ s, float4* __restrict__ d, int n4) {
  int i = blockIdx.x * blockDim.x + threadIdx.x;
  if (i < n4) d[i] = s[i];
}

__global__ void zero_kernel(float* __restrict__ p, int n) {
  int i = blockIdx.x * blockDim.x + threadIdx.x;
  if (i < n) p[i] = 0.0f;
}

// ---------------------------------------------------------------------------
// Fused GINE edge stage: e = edge_attr @ We + be ; m = relu(x[src] + e) ;
// aggr[dst] += m   (aggr pre-initialized to x, so aggr ends as x + scatter-sum)
// One wave handles 16 edges x 128 outputs: 8 N-tiles x 2 WMMAs (K=64).
// Packed weights (16 KB) staged block-wide in LDS via async global->LDS DMA.
// Grid is exact: 6250 blocks * 8 waves * 16 edges = 800000.
// ---------------------------------------------------------------------------
__global__ __launch_bounds__(256) void edge_msg_kernel(
    const float* __restrict__ xin,        // [N_NODES,128] gather source
    const float* __restrict__ eattr,      // [N_EDGES,64]
    const int*   __restrict__ eidx,       // [2,N_EDGES]
    const _Float16* __restrict__ wpack,   // packed [64x128] f16
    const float* __restrict__ bias,       // [128]
    float* __restrict__ aggr)             // [N_NODES,128] accumulator
{
  __shared__ __align__(32) _Float16 wlds[EDGE_IN * HIDDEN];   // 16 KB

  // --- stage packed weights into LDS once per block ---
#if HAS_ASYNC_LDS
  {
    typedef __attribute__((ext_vector_type(4))) int i4;
    typedef __attribute__((address_space(1))) i4 gi4;
    typedef __attribute__((address_space(3))) i4 li4;
    gi4* g = (gi4*)wpack;                 // 16-byte granules, global
    li4* l = (li4*)wlds;                  // 16-byte granules, LDS
    const int t = threadIdx.x;
#pragma unroll
    for (int it = 0; it < 4; ++it)
      __builtin_amdgcn_global_load_async_to_lds_b128(g + t + it * 256,
                                                     l + t + it * 256, 0, 0);
    __builtin_amdgcn_s_wait_asynccnt(0);
  }
#else
  {
    const float4* g4 = (const float4*)wpack;
    float4* l4 = (float4*)wlds;
#pragma unroll
    for (int it = 0; it < 4; ++it)
      l4[threadIdx.x + it * 256] = g4[threadIdx.x + it * 256];
  }
#endif
  __syncthreads();

  const int lane = threadIdx.x & 31;
  const int tile = blockIdx.x * 8 + (threadIdx.x >> 5);
  const int e0 = tile * 16;
  const int ln = lane & 15;
  const int hi = lane >> 4;

  // A fragments: 16 edge rows x 64 K, f32 -> f16 on the fly
  const float* ap = eattr + (size_t)(e0 + ln) * EDGE_IN;
  __builtin_prefetch(ap + 16 * EDGE_IN, 0, 0);   // next tile's rows
  v16h a0, a1;
#pragma unroll
  for (int t = 0; t < 16; ++t) {
    a0[t] = (_Float16)ap[kmap(t, hi)];
    a1[t] = (_Float16)ap[32 + kmap(t, hi)];
  }

  // 32-bit element offsets (tables < 2^31 elements) -> saddr+voffset addressing
  int so[8], dofs[8];
#pragma unroll
  for (int r = 0; r < 8; ++r) {
    int e = e0 + r + hi * 8;            // C/D layout: VGPR r -> M = r (+8 for hi lanes)
    so[r]   = eidx[e] * HIDDEN + ln;
    dofs[r] = eidx[N_EDGES + e] * HIDDEN + ln;
  }

  const v16h* B = (const v16h*)wlds;
#pragma unroll
  for (int nt = 0; nt < 8; ++nt) {
    const int n0 = nt * 16;
    v16h b0 = B[(nt * 2 + 0) * 32 + lane];
    v16h b1 = B[(nt * 2 + 1) * 32 + lane];
    v8f c = {};
    c = wmma_f16(a0, b0, c);
    c = wmma_f16(a1, b1, c);

    // batch the 8 independent gathers first so they issue as one clause and
    // their latency overlaps the adds/atomics (instead of load->wait->atomic
    // serialization per element)
    float xg[8];
#pragma unroll
    for (int r = 0; r < 8; ++r) xg[r] = xin[so[r] + n0];

    float bv = bias[n0 + ln];
#pragma unroll
    for (int r = 0; r < 8; ++r) {
      float v = c[r] + bv + xg[r];
      v = v > 0.0f ? v : 0.0f;
      atomicAdd(&aggr[dofs[r] + n0], v);
    }
  }
}

// ---------------------------------------------------------------------------
// Node MLP: out = relu( relu(h @ W1 + b1) @ W2 + b2 )
// One wave per 16-row tile; intermediate staged as f16 in LDS. 64 WMMAs/wave.
// ---------------------------------------------------------------------------
__global__ __launch_bounds__(128) void node_mlp_kernel(
    const float* __restrict__ hin,
    const _Float16* __restrict__ w1p, const float* __restrict__ b1,
    const _Float16* __restrict__ w2p, const float* __restrict__ b2,
    float* __restrict__ hout)
{
  __shared__ __align__(32) _Float16 tbuf[4][16 * HIDDEN];
  const int lane = threadIdx.x & 31;
  const int wv = threadIdx.x >> 5;
  const int NT = N_NODES / 16;          // 3125
  int tile = blockIdx.x * 4 + wv;
  if (tile >= NT) tile = NT - 1;        // duplicate work, same values (benign)
  _Float16* T = tbuf[wv];
  const int ln = lane & 15, hi = lane >> 4;
  const int row0 = tile * 16;

  v16h A[4];
  const float* ap = hin + (row0 + ln) * HIDDEN;   // 32-bit offset
#pragma unroll
  for (int kk = 0; kk < 4; ++kk)
#pragma unroll
    for (int t = 0; t < 16; ++t)
      A[kk][t] = (_Float16)ap[kk * 32 + kmap(t, hi)];

  const v16h* B1 = (const v16h*)w1p;
#pragma unroll
  for (int nt = 0; nt < 8; ++nt) {
    v8f c = {};
#pragma unroll
    for (int kk = 0; kk < 4; ++kk)
      c = wmma_f16(A[kk], B1[(nt * 4 + kk) * 32 + lane], c);
    float bv = b1[nt * 16 + ln];
#pragma unroll
    for (int r = 0; r < 8; ++r) {
      float v = c[r] + bv;
      T[(r + hi * 8) * HIDDEN + nt * 16 + ln] = (_Float16)(v > 0.0f ? v : 0.0f);
    }
  }
  __syncthreads();

  v16h A2[4];
#pragma unroll
  for (int kk = 0; kk < 4; ++kk)
#pragma unroll
    for (int t = 0; t < 16; ++t)
      A2[kk][t] = T[ln * HIDDEN + kk * 32 + kmap(t, hi)];

  const int ho = (row0 + hi * 8) * HIDDEN + ln;   // 32-bit output base offset
  const v16h* B2 = (const v16h*)w2p;
#pragma unroll
  for (int nt = 0; nt < 8; ++nt) {
    v8f c = {};
#pragma unroll
    for (int kk = 0; kk < 4; ++kk)
      c = wmma_f16(A2[kk], B2[(nt * 4 + kk) * 32 + lane], c);
    float bv = b2[nt * 16 + ln];
#pragma unroll
    for (int r = 0; r < 8; ++r) {
      float v = c[r] + bv;
      hout[ho + r * HIDDEN + nt * 16] = v > 0.0f ? v : 0.0f;
    }
  }
}

// ---------------------------------------------------------------------------
// Global mean pool (atomics into 512x128 sums, hot in L2) + counts
// ---------------------------------------------------------------------------
__global__ void pool_kernel(const float* __restrict__ h, const int* __restrict__ batch,
                            float* __restrict__ sums, float* __restrict__ cnt) {
  int i = blockIdx.x * blockDim.x + threadIdx.x;
  if (i >= N_NODES * HIDDEN) return;
  int node = i >> 7;
  int col  = i & 127;
  int g = batch[node];
  atomicAdd(&sums[g * HIDDEN + col], h[i]);
  if (col == 0) atomicAdd(&cnt[g], 1.0f);
}

// ---------------------------------------------------------------------------
// Heads: s = relu(mean @ fc_w + fc_b); three 64-dot heads -> out[3*512]
// ---------------------------------------------------------------------------
__global__ __launch_bounds__(64) void head_kernel(
    const float* __restrict__ sums, const float* __restrict__ cnt,
    const float* __restrict__ fcw, const float* __restrict__ fcb,
    const float* __restrict__ hSw, const float* __restrict__ hSb,
    const float* __restrict__ hPw, const float* __restrict__ hPb,
    const float* __restrict__ hNw, const float* __restrict__ hNb,
    float* __restrict__ out)
{
  __shared__ float s[SHARED];
  int g = blockIdx.x;
  int j = threadIdx.x;
  float c = cnt[g];
  c = c > 1.0f ? c : 1.0f;
  float acc = fcb[j];
#pragma unroll 4
  for (int k = 0; k < HIDDEN; ++k)
    acc += (sums[g * HIDDEN + k] / c) * fcw[k * SHARED + j];
  s[j] = acc > 0.0f ? acc : 0.0f;
  __syncthreads();
  if (j < 3) {
    const float* w = (j == 0) ? hSw : (j == 1) ? hPw : hNw;
    float a = (j == 0) ? hSb[0] : (j == 1) ? hPb[0] : hNb[0];
    for (int k = 0; k < SHARED; ++k) a += s[k] * w[k];
    out[j * N_GRAPHS + g] = a;
  }
}

// ---------------------------------------------------------------------------
extern "C" void kernel_launch(void* const* d_in, const int* in_sizes, int n_in,
                              void* d_out, int out_size, void* d_ws, size_t ws_size,
                              hipStream_t stream) {
  const float* x     = (const float*)d_in[0];
  const int*   eidx  = (const int*)d_in[1];
  const float* eattr = (const float*)d_in[2];
  const int*   batch = (const int*)d_in[3];
  const float* e1_w  = (const float*)d_in[4];
  const float* e1_b  = (const float*)d_in[5];
  const float* n1_w1 = (const float*)d_in[6];
  const float* n1_b1 = (const float*)d_in[7];
  const float* n1_w2 = (const float*)d_in[8];
  const float* n1_b2 = (const float*)d_in[9];
  const float* e2_w  = (const float*)d_in[10];
  const float* e2_b  = (const float*)d_in[11];
  const float* n2_w1 = (const float*)d_in[12];
  const float* n2_b1 = (const float*)d_in[13];
  const float* n2_w2 = (const float*)d_in[14];
  const float* n2_b2 = (const float*)d_in[15];
  const float* fc_w  = (const float*)d_in[16];
  const float* fc_b  = (const float*)d_in[17];
  const float* hS_w  = (const float*)d_in[18];
  const float* hS_b  = (const float*)d_in[19];
  const float* hP_w  = (const float*)d_in[20];
  const float* hP_b  = (const float*)d_in[21];
  const float* hN_w  = (const float*)d_in[22];
  const float* hN_b  = (const float*)d_in[23];
  float* out = (float*)d_out;

  char* ws = (char*)d_ws;
  size_t off = 0;
  const size_t nodeBytes = (size_t)N_NODES * HIDDEN * sizeof(float);
  float* bufA = (float*)(ws + off); off += nodeBytes;           // aggr accumulator / MLP input
  float* bufB = (float*)(ws + off); off += nodeBytes;           // layer output
  _Float16* pe1  = (_Float16*)(ws + off); off += (size_t)EDGE_IN * HIDDEN * 2;
  _Float16* pw11 = (_Float16*)(ws + off); off += (size_t)HIDDEN  * HIDDEN * 2;
  _Float16* pw12 = (_Float16*)(ws + off); off += (size_t)HIDDEN  * HIDDEN * 2;
  _Float16* pe2  = (_Float16*)(ws + off); off += (size_t)EDGE_IN * HIDDEN * 2;
  _Float16* pw21 = (_Float16*)(ws + off); off += (size_t)HIDDEN  * HIDDEN * 2;
  _Float16* pw22 = (_Float16*)(ws + off); off += (size_t)HIDDEN  * HIDDEN * 2;
  float* sums = (float*)(ws + off); off += (size_t)N_GRAPHS * HIDDEN * sizeof(float);
  float* cnt  = (float*)(ws + off); off += (size_t)N_GRAPHS * sizeof(float);

  // Pack weights to f16 WMMA-B fragment order (tiny; once per launch)
  pack_w_kernel<<<(EDGE_IN * HIDDEN + 255) / 256, 256, 0, stream>>>(e1_w, pe1, EDGE_IN, HIDDEN);
  pack_w_kernel<<<(HIDDEN * HIDDEN + 255) / 256, 256, 0, stream>>>(n1_w1, pw11, HIDDEN, HIDDEN);
  pack_w_kernel<<<(HIDDEN * HIDDEN + 255) / 256, 256, 0, stream>>>(n1_w2, pw12, HIDDEN, HIDDEN);
  pack_w_kernel<<<(EDGE_IN * HIDDEN + 255) / 256, 256, 0, stream>>>(e2_w, pe2, EDGE_IN, HIDDEN);
  pack_w_kernel<<<(HIDDEN * HIDDEN + 255) / 256, 256, 0, stream>>>(n2_w1, pw21, HIDDEN, HIDDEN);
  pack_w_kernel<<<(HIDDEN * HIDDEN + 255) / 256, 256, 0, stream>>>(n2_w2, pw22, HIDDEN, HIDDEN);

  const int n4 = N_NODES * HIDDEN / 4;
  const int edgeBlocks = N_EDGES / 16 / 8;            // 6250 (exact)
  const int nodeBlocks = (N_NODES / 16 + 3) / 4;      // 782

  // ---- Layer 1 ----
  copy4_kernel<<<(n4 + 255) / 256, 256, 0, stream>>>((const float4*)x, (float4*)bufA, n4);
  edge_msg_kernel<<<edgeBlocks, 256, 0, stream>>>(x, eattr, eidx, pe1, e1_b, bufA);
  node_mlp_kernel<<<nodeBlocks, 128, 0, stream>>>(bufA, pw11, n1_b1, pw12, n1_b2, bufB);

  // ---- Layer 2 ----
  copy4_kernel<<<(n4 + 255) / 256, 256, 0, stream>>>((const float4*)bufB, (float4*)bufA, n4);
  edge_msg_kernel<<<edgeBlocks, 256, 0, stream>>>(bufB, eattr, eidx, pe2, e2_b, bufA);
  node_mlp_kernel<<<nodeBlocks, 128, 0, stream>>>(bufA, pw21, n2_b1, pw22, n2_b2, bufB);

  // ---- Pool + heads ----
  const int zn = N_GRAPHS * HIDDEN + N_GRAPHS;        // sums and cnt are contiguous
  zero_kernel<<<(zn + 255) / 256, 256, 0, stream>>>(sums, zn);
  pool_kernel<<<(N_NODES * HIDDEN + 255) / 256, 256, 0, stream>>>(bufB, batch, sums, cnt);
  head_kernel<<<N_GRAPHS, 64, 0, stream>>>(sums, cnt, fc_w, fc_b, hS_w, hS_b,
                                           hP_w, hP_b, hN_w, hN_b, out);
}
